// GptOssAttention_63307817943055
// MI455X (gfx1250) — matrix operand order repaired and checked by reading, not compiled, not run on hardware
//
#include <hip/hip_runtime.h>

typedef __attribute__((ext_vector_type(16))) _Float16 v16h;
typedef __attribute__((ext_vector_type(8)))  float    v8f;

#define S_LEN   2048
#define HIDDEN  2880
#define NH      64
#define NKV     8
#define HD      64
#define QDIM    (NH * HD)    /* 4096 */
#define KVDIM   (NKV * HD)   /* 512  */
#define WINDOW  128
#define QK_SCALE 0.125f
#define NEG_INF (-1e30f)

// ---------------------------------------------------------------------------
// f32 -> f16 conversion (inputs are O(1) scaled; f16 + f32-accum WMMA is fine)
// ---------------------------------------------------------------------------
__global__ void f32_to_f16(const float* __restrict__ src,
                           _Float16* __restrict__ dst, int n) {
  int i = blockIdx.x * blockDim.x + threadIdx.x;
  if (i < n) dst[i] = (_Float16)src[i];
}

// ---------------------------------------------------------------------------
// RoPE for Q (rotate-half), f32 in -> f16 out, layout preserved [S][H*D].
// ---------------------------------------------------------------------------
__global__ void rope_q_kernel(const float* __restrict__ x,
                              const float* __restrict__ cosb,
                              const float* __restrict__ sinb,
                              _Float16* __restrict__ out, int n) {
  int i = blockIdx.x * blockDim.x + threadIdx.x;
  if (i >= n) return;
  int s = i / QDIM;
  int d = i % HD;
  float c  = cosb[s * HD + d];
  float sn = sinb[s * HD + d];
  float v  = x[i];
  float vp = (d < 32) ? x[i + 32] : x[i - 32];
  float r  = (d < 32) ? (v * c - vp * sn) : (v * c + vp * sn);
  out[i] = (_Float16)r;
}

// ---------------------------------------------------------------------------
// RoPE for K, fused with transpose:  kf[S][KV*D] (f32) -> kT[KV*D][S] (f16).
// Transposed layout puts consecutive keys in consecutive lanes for the K^T B-fragment loads.
// ---------------------------------------------------------------------------
__global__ void rope_k_t_kernel(const float* __restrict__ kf,
                                const float* __restrict__ cosb,
                                const float* __restrict__ sinb,
                                _Float16* __restrict__ kT) {
  int idx = blockIdx.x * blockDim.x + threadIdx.x;  // over KVDIM * S_LEN
  int row = idx >> 11;    // kv_head*64 + d
  int s   = idx & (S_LEN - 1);
  int d   = row & (HD - 1);
  float c  = cosb[s * HD + d];
  float sn = sinb[s * HD + d];
  float v  = kf[(size_t)s * KVDIM + row];
  int prow = (d < 32) ? row + 32 : row - 32;
  float vp = kf[(size_t)s * KVDIM + prow];
  float r  = (d < 32) ? (v * c - vp * sn) : (v * c + vp * sn);
  kT[idx] = (_Float16)r;
}

// ---------------------------------------------------------------------------
// Tiled WMMA GEMM:  C(MxN) = A(MxK) @ B(KxN) + bias,  A/B f16 row-major.
// Block = 256 threads = 8 waves; tile 128(M) x 64(N), K-step 32.
// Global -> LDS staging uses CDNA5 async loads (ASYNCcnt) with double
// buffering so HBM traffic overlaps the WMMA pipeline.
// ---------------------------------------------------------------------------
template <int STORE_F16>
__global__ __launch_bounds__(256) void wmma_gemm_bias(
    const _Float16* __restrict__ A, const _Float16* __restrict__ B,
    const float* __restrict__ bias, float* __restrict__ Cf,
    _Float16* __restrict__ Ch, int M, int N, int K) {
  __shared__ __align__(16) _Float16 lA[2][128 * 40];  // 32 cols + 8 pad halves
  __shared__ __align__(16) _Float16 lB[2][32 * 72];   // 64 cols + 8 pad halves

  const int tid  = threadIdx.x;
  const int lane = tid & 31;
  const int wave = tid >> 5;
  const int m0   = blockIdx.y * 128;
  const int n0   = blockIdx.x * 64;

  // staging coordinates (A: 512 uint4 -> 2/thread, B: 256 uint4 -> 1/thread)
  const int ar0 = tid >> 2,         ac0 = (tid & 3) * 8;
  const int ar1 = (tid + 256) >> 2, ac1 = ((tid + 256) & 3) * 8;
  const int br  = tid >> 3,         bc  = (tid & 7) * 8;

  const _Float16* gA0 = A + (size_t)(m0 + ar0) * K + ac0;
  const _Float16* gA1 = A + (size_t)(m0 + ar1) * K + ac1;
  const _Float16* gB0 = B + (size_t)br * N + n0 + bc;

  auto issue_tile = [&](int buf, int k0) {
    unsigned la0 = (unsigned)(size_t)&lA[buf][ar0 * 40 + ac0];
    unsigned la1 = (unsigned)(size_t)&lA[buf][ar1 * 40 + ac1];
    unsigned lb0 = (unsigned)(size_t)&lB[buf][br * 72 + bc];
    unsigned long long ga0 = (unsigned long long)(size_t)(gA0 + k0);
    unsigned long long ga1 = (unsigned long long)(size_t)(gA1 + k0);
    unsigned long long gb0 = (unsigned long long)(size_t)(gB0 + (size_t)k0 * N);
    asm volatile("global_load_async_to_lds_b128 %0, %1, off"
                 :: "v"(la0), "v"(ga0) : "memory");
    asm volatile("global_load_async_to_lds_b128 %0, %1, off"
                 :: "v"(la1), "v"(ga1) : "memory");
    asm volatile("global_load_async_to_lds_b128 %0, %1, off"
                 :: "v"(lb0), "v"(gb0) : "memory");
  };

  v8f acc[4] = {};
  const int lm = lane & 15, hs = lane >> 4;
  const int mrow = wave * 16 + lm;

  issue_tile(0, 0);
  int buf = 0;
  for (int k0 = 0; k0 < K; k0 += 32) {
    asm volatile("s_wait_asynccnt 0x0" ::: "memory");
    __syncthreads();
    if (k0 + 32 < K) issue_tile(buf ^ 1, k0 + 32);

    const _Float16* pA = lA[buf];
    const _Float16* pB = lB[buf];

    // A fragment (16x32 f16): lane m = lm, halves 0..7 -> k = hs*8..,
    // halves 8..15 -> k = 16 + hs*8..
    v16h af;
    *((uint4*)&af)       = *(const uint4*)(pA + mrow * 40 + hs * 8);
    *(((uint4*)&af) + 1) = *(const uint4*)(pA + mrow * 40 + 16 + hs * 8);

#pragma unroll
    for (int nt = 0; nt < 4; ++nt) {
      // B fragment (32x16): lane <-> N col, halves = K rows (same k order as A)
      v16h bf;
      for (int s = 0; s < 16; ++s)
        bf[s] = pB[((s < 8) ? hs * 8 + s : 16 + hs * 8 + (s - 8)) * 72 + nt * 16 + lm];
      acc[nt] = __builtin_amdgcn_wmma_f32_16x16x32_f16(
          false, af, false, bf, (short)0, acc[nt], false, false);
    }
    buf ^= 1;
  }

  // epilogue: C layout lane n = lm, VGPR r -> row r (lanes<16) / 8+r
#pragma unroll
  for (int nt = 0; nt < 4; ++nt) {
    const int col = n0 + nt * 16 + lm;
    const float bv = bias[col];
#pragma unroll
    for (int r = 0; r < 8; ++r) {
      const int row = m0 + wave * 16 + r + hs * 8;
      const float v = acc[nt][r] + bv;
      if (STORE_F16)
        Ch[(size_t)row * N + col] = (_Float16)v;
      else
        Cf[(size_t)row * N + col] = v;
    }
  }
}

// ---------------------------------------------------------------------------
// Sliding-window flash attention with sinks.  One wave per (16-query tile,
// head).  Key chunks of 32: scores = 4 WMMA, PV = 4 WMMA per chunk.
// K is consumed from the transposed kT[KV*D][S] layout -> per-element
// fragment loads, S_LEN-strided per lane, consecutive keys across lanes.
// ---------------------------------------------------------------------------
__global__ __launch_bounds__(128) void attn_kernel(
    const _Float16* __restrict__ qh, const _Float16* __restrict__ kT,
    const _Float16* __restrict__ vh, const float* __restrict__ sinks,
    _Float16* __restrict__ outh) {
  __shared__ __align__(16) _Float16 pbuf[4][16 * 40];  // P transpose, per wave

  const int lane = threadIdx.x & 31;
  const int wv   = threadIdx.x >> 5;
  const int gw   = blockIdx.x * 4 + wv;
  const int qt   = gw >> 6;   // query tile 0..127
  const int h    = gw & 63;   // head
  const int kvh  = h >> 3;    // GQA group
  const int i0   = qt * 16;
  const int lm   = lane & 15, hs = lane >> 4;

  // Q fragments (16x64 -> two 16x32 A-frags), direct from global
  const _Float16* qbase = qh + (size_t)(i0 + lm) * QDIM + h * HD;
  v16h qa0, qa1;
  *((uint4*)&qa0)       = *(const uint4*)(qbase + hs * 8);
  *(((uint4*)&qa0) + 1) = *(const uint4*)(qbase + 16 + hs * 8);
  *((uint4*)&qa1)       = *(const uint4*)(qbase + 32 + hs * 8);
  *(((uint4*)&qa1) + 1) = *(const uint4*)(qbase + 48 + hs * 8);

  // K^T column for this lane: key = lm; kd rows 0..31 (low) / 32..63 (high)
  const _Float16* kTl = kT + (size_t)(kvh * HD) * S_LEN + lm;
  const _Float16* kTh = kT + (size_t)(kvh * HD + 32) * S_LEN + lm;

  float mrun[8], lrun[8];
#pragma unroll
  for (int r = 0; r < 8; ++r) { mrun[r] = NEG_INF; lrun[r] = 0.f; }
  v8f o[4] = {};

  int lowj = i0 - (WINDOW - 1);
  if (lowj < 0) lowj = 0;
  const int c0 = lowj >> 5;
  const int c1 = (i0 + 15) >> 5;

  for (int c = c0; c <= c1; ++c) {
    const int kb = c * 32;

    // scores S(16q x 32k)
    v8f sa[2] = {};
#pragma unroll
    for (int nt = 0; nt < 2; ++nt) {
      // K^T B-frags: lane <-> key col, halves = head-dim rows (same k order as Q)
      v16h klo, khi;
      for (int s = 0; s < 16; ++s) {
        const int kd = (s < 8) ? hs * 8 + s : 16 + hs * 8 + (s - 8);
        klo[s] = kTl[(size_t)kd * S_LEN + kb + nt * 16];
        khi[s] = kTh[(size_t)kd * S_LEN + kb + nt * 16];
      }
      sa[nt] = __builtin_amdgcn_wmma_f32_16x16x32_f16(
          false, qa0, false, klo, (short)0, sa[nt], false, false);
      sa[nt] = __builtin_amdgcn_wmma_f32_16x16x32_f16(
          false, qa1, false, khi, (short)0, sa[nt], false, false);
    }

    // online softmax (rows live in (VGPR r, lane-half); cols in 16 lanes)
    float p[2][8], alpha[8];
#pragma unroll
    for (int r = 0; r < 8; ++r) {
      const int i = i0 + r + hs * 8;
      float rowm = NEG_INF;
#pragma unroll
      for (int nt = 0; nt < 2; ++nt) {
        const int j = kb + nt * 16 + lm;
        const bool ok = (j <= i) && (j > i - WINDOW);
        const float sc = ok ? sa[nt][r] * QK_SCALE : NEG_INF;
        p[nt][r] = sc;
        rowm = fmaxf(rowm, sc);
      }
#pragma unroll
      for (int off = 1; off < 16; off <<= 1)
        rowm = fmaxf(rowm, __shfl_xor(rowm, off, 32));
      const float mnew = fmaxf(mrun[r], rowm);
      alpha[r] = __expf(mrun[r] - mnew);
      float rsum = 0.f;
#pragma unroll
      for (int nt = 0; nt < 2; ++nt) {
        const int j = kb + nt * 16 + lm;
        const bool ok = (j <= i) && (j > i - WINDOW);
        const float pe = ok ? __expf(p[nt][r] - mnew) : 0.f;
        p[nt][r] = pe;
        rsum += pe;
      }
#pragma unroll
      for (int off = 1; off < 16; off <<= 1)
        rsum += __shfl_xor(rsum, off, 32);
      lrun[r] = lrun[r] * alpha[r] + rsum;
      mrun[r] = mnew;
    }
#pragma unroll
    for (int f = 0; f < 4; ++f)
#pragma unroll
      for (int r = 0; r < 8; ++r) o[f][r] *= alpha[r];

    // transpose P through LDS (wave-private; same-wave DS ordering)
#pragma unroll
    for (int r = 0; r < 8; ++r) {
      const int mr = r + hs * 8;
      pbuf[wv][mr * 40 + lm]      = (_Float16)p[0][r];
      pbuf[wv][mr * 40 + 16 + lm] = (_Float16)p[1][r];
    }
    asm volatile("s_wait_dscnt 0x0" ::: "memory");
    v16h pa;
    *((uint4*)&pa)       = *(const uint4*)(&pbuf[wv][lm * 40 + hs * 8]);
    *(((uint4*)&pa) + 1) = *(const uint4*)(&pbuf[wv][lm * 40 + 16 + hs * 8]);

    // O += P @ V   (V B-frag: lane <-> dim col, halves = key rows, same k order as P)
#pragma unroll
    for (int f = 0; f < 4; ++f) {
      const _Float16* vp = vh + (size_t)kb * KVDIM + kvh * HD + f * 16 + lm;
      v16h vb;
      for (int s = 0; s < 16; ++s)
        vb[s] = vp[(size_t)((s < 8) ? hs * 8 + s : 16 + hs * 8 + (s - 8)) * KVDIM];
      o[f] = __builtin_amdgcn_wmma_f32_16x16x32_f16(
          false, pa, false, vb, (short)0, o[f], false, false);
    }
  }

  // fold in sink logit, normalize, store f16 for the output projection
  const float sk = sinks[h];
#pragma unroll
  for (int r = 0; r < 8; ++r) {
    const float m2    = fmaxf(mrun[r], sk);
    const float beta  = __expf(mrun[r] - m2);
    const float denom = lrun[r] * beta + __expf(sk - m2);
    const float scl   = beta / denom;
    const int row = i0 + r + hs * 8;
#pragma unroll
    for (int f = 0; f < 4; ++f)
      outh[(size_t)row * QDIM + h * HD + f * 16 + lm] =
          (_Float16)(o[f][r] * scl);
  }
}

// ---------------------------------------------------------------------------
extern "C" void kernel_launch(void* const* d_in, const int* in_sizes, int n_in,
                              void* d_out, int out_size, void* d_ws,
                              size_t ws_size, hipStream_t stream) {
  const float* hid   = (const float*)d_in[0];
  const float* cosb  = (const float*)d_in[1];
  const float* sinb  = (const float*)d_in[2];
  const float* Wq    = (const float*)d_in[3];
  const float* bq    = (const float*)d_in[4];
  const float* Wk    = (const float*)d_in[5];
  const float* bk    = (const float*)d_in[6];
  const float* Wv    = (const float*)d_in[7];
  const float* bv    = (const float*)d_in[8];
  const float* Wo    = (const float*)d_in[9];
  const float* bo    = (const float*)d_in[10];
  const float* sinks = (const float*)d_in[11];
  float* out = (float*)d_out;

  char* w = (char*)d_ws;
  auto carve = [&](size_t bytes) {
    char* p = w;
    w += (bytes + 255) & ~(size_t)255;
    return p;
  };
  _Float16* Xh  = (_Float16*)carve((size_t)S_LEN * HIDDEN * 2);
  _Float16* Wqh = (_Float16*)carve((size_t)HIDDEN * QDIM * 2);
  _Float16* Wkh = (_Float16*)carve((size_t)HIDDEN * KVDIM * 2);
  _Float16* Wvh = (_Float16*)carve((size_t)HIDDEN * KVDIM * 2);
  _Float16* Woh = (_Float16*)carve((size_t)QDIM * HIDDEN * 2);
  float*    qf  = (float*)   carve((size_t)S_LEN * QDIM * 4);
  float*    kf  = (float*)   carve((size_t)S_LEN * KVDIM * 4);
  _Float16* qh2 = (_Float16*)carve((size_t)S_LEN * QDIM * 2);
  _Float16* kT  = (_Float16*)carve((size_t)S_LEN * KVDIM * 2);
  _Float16* vh2 = (_Float16*)carve((size_t)S_LEN * KVDIM * 2);
  _Float16* ah  = (_Float16*)qf;  // alias: qf is dead after RoPE

  int n;
  n = S_LEN * HIDDEN;
  f32_to_f16<<<(n + 255) / 256, 256, 0, stream>>>(hid, Xh, n);
  n = HIDDEN * QDIM;
  f32_to_f16<<<(n + 255) / 256, 256, 0, stream>>>(Wq, Wqh, n);
  n = HIDDEN * KVDIM;
  f32_to_f16<<<(n + 255) / 256, 256, 0, stream>>>(Wk, Wkh, n);
  f32_to_f16<<<(n + 255) / 256, 256, 0, stream>>>(Wv, Wvh, n);
  n = QDIM * HIDDEN;
  f32_to_f16<<<(n + 255) / 256, 256, 0, stream>>>(Wo, Woh, n);

  // QKV projections
  wmma_gemm_bias<0><<<dim3(QDIM / 64, S_LEN / 128), 256, 0, stream>>>(
      Xh, Wqh, bq, qf, nullptr, S_LEN, QDIM, HIDDEN);
  wmma_gemm_bias<0><<<dim3(KVDIM / 64, S_LEN / 128), 256, 0, stream>>>(
      Xh, Wkh, bk, kf, nullptr, S_LEN, KVDIM, HIDDEN);
  wmma_gemm_bias<1><<<dim3(KVDIM / 64, S_LEN / 128), 256, 0, stream>>>(
      Xh, Wvh, bv, nullptr, vh2, S_LEN, KVDIM, HIDDEN);

  // RoPE (K fused with transpose to kT[KV*D][S])
  n = S_LEN * QDIM;
  rope_q_kernel<<<n / 256, 256, 0, stream>>>(qf, cosb, sinb, qh2, n);
  n = S_LEN * KVDIM;
  rope_k_t_kernel<<<n / 256, 256, 0, stream>>>(kf, cosb, sinb, kT);

  // attention: 128 qtiles * 64 heads = 8192 waves, 4 waves/block
  attn_kernel<<<2048, 128, 0, stream>>>(qh2, kT, vh2, sinks, ah);

  // output projection -> f32 d_out
  wmma_gemm_bias<0><<<dim3(HIDDEN / 64, S_LEN / 128), 256, 0, stream>>>(
      ah, Woh, bo, out, nullptr, S_LEN, HIDDEN, QDIM);
}
